// Model_28389733826561
// MI455X (gfx1250) — compile-verified, matrix operand order checked by reading, not run on hardware
//
#include <hip/hip_runtime.h>

typedef __bf16 bf16;
typedef __attribute__((ext_vector_type(16))) __bf16 v16bf;
typedef __attribute__((ext_vector_type(8)))  __bf16 v8bf;
typedef __attribute__((ext_vector_type(8)))  float   v8f;

#define BATCH   256
#define TSTEPS  10
#define WIDTH0  256
#define W1      86
#define W2      29
#define W3      10
#define HID     256
#define OUTF    64
#define PREDN   9
#define EPSV    1e-12f

union FragBF { v16bf v; v8bf h[2]; };

__device__ __forceinline__ float sigf(float x) { return 1.0f / (1.0f + expf(-x)); }

// ---------------------------------------------------------------------------
// Generic bf16 WMMA GEMM (used for conv2/conv3 im2col GEMMs and feats@Wx):
//   A: bf16 row-major (M x K), BT: bf16 row-major (N x K)
//   out = acc + bias?[n] + Cinit?[m,n]
// One wave per 16x16 tile, 4 waves/block across N. M%16==0, K%32==0.
// ---------------------------------------------------------------------------
__global__ void gemm_bf16_wmma(const bf16* __restrict__ A, const bf16* __restrict__ BT,
                               const float* __restrict__ Cinit, int ldCin,
                               const float* __restrict__ bias,
                               float* __restrict__ Cout, int ldCout,
                               int M, int N, int K)
{
    const int lane = threadIdx.x;
    const int wave = threadIdx.y;
    const int n0 = (blockIdx.x * 4 + wave) * 16;
    const int m0 = blockIdx.y * 16;
    if (n0 >= N) return;                          // wave-uniform

    const int half = lane >> 4;
    const int l16  = lane & 15;
    const bf16* pa = A  + (size_t)(m0 + l16) * K + half * 8;
    const bf16* pb = BT + (size_t)(n0 + l16) * K + half * 8;

    v8f acc = {0.f, 0.f, 0.f, 0.f, 0.f, 0.f, 0.f, 0.f};
    for (int k0 = 0; k0 < K; k0 += 32) {
        FragBF a, b;
        a.h[0] = *(const v8bf*)(pa);
        a.h[1] = *(const v8bf*)(pa + 16);
        b.h[0] = *(const v8bf*)(pb);
        b.h[1] = *(const v8bf*)(pb + 16);
        pa += 32; pb += 32;
        acc = __builtin_amdgcn_wmma_f32_16x16x32_bf16(
            false, a.v, false, b.v, (short)0, acc, false, false);
    }

    const int n = n0 + l16;
    const float bn = bias ? bias[n] : 0.f;
    #pragma unroll
    for (int v = 0; v < 8; ++v) {
        const int m = m0 + v + half * 8;
        float o = acc[v] + bn + (Cinit ? Cinit[(size_t)m * ldCin + n] : 0.f);
        Cout[(size_t)m * ldCout + n] = o;
    }
}

// ---------------------------------------------------------------------------
// Fully fused recurrence: one workgroup = 16 batch rows through the ENTIRE
// T-loop (LSTM gates + 9x(3 Euler steps) ODE + softmax head).
//   - h (bf16), c (f32), y/u bf16 ping-pong operand buffers in LDS
//   - y f32 state lives in wave registers across all Euler steps of a t
//   - A-fragments via ds_load_b128 from LDS; B (weights) stream from L2
// Block = 256 threads (8 waves); wave w owns output columns [32w, 32w+32).
// ---------------------------------------------------------------------------
__global__ __launch_bounds__(256) void recurrent_fused(
    const float* __restrict__ xproj,   // (B*T, 1024), bias blstm already added
    const bf16* __restrict__ WhT,      // (1024, 256)
    const bf16* __restrict__ Wf1T,     // (256, 256)
    const bf16* __restrict__ Wf2T,     // (256, 256)
    const bf16* __restrict__ WfcT,     // (64, 256)
    const float* __restrict__ bf1, const float* __restrict__ bf2,
    const float* __restrict__ bfc,
    float* __restrict__ out, float dt)
{
    __shared__ float cstate[16][HID];       // 16 KB  LSTM cell state
    __shared__ bf16  hbuf[16][HID];         //  8 KB  LSTM hidden (GEMM A operand)
    __shared__ bf16  abufA[16][HID];        //  8 KB  ping
    __shared__ bf16  abufB[16][HID];        //  8 KB  pong
    __shared__ float lbuf[16][OUTF];        //  4 KB  logits
    __shared__ float srow[16][2];           //        softmax max / inv-sum

    const int tid  = threadIdx.x;
    const int lane = tid & 31;
    const int wave = tid >> 5;
    const int half = lane >> 4;
    const int l16  = lane & 15;
    const int bbase = blockIdx.x * 16;

    // zero persistent state
    for (int i = tid; i < 16 * HID; i += 256) {
        ((float*)cstate)[i] = 0.f;
        ((bf16*)hbuf)[i] = (bf16)0.f;
    }
    __syncthreads();

    // K-loop helper: 2 output tiles (cols (2w..2w+1)*16), A from LDS, B global
    auto kloop2 = [&](const bf16 (&A)[16][HID], const bf16* __restrict__ BT,
                      v8f acc[2]) {
        for (int k0 = 0; k0 < HID; k0 += 32) {
            FragBF a;
            a.h[0] = *(const v8bf*)&A[l16][k0 + half * 8];
            a.h[1] = *(const v8bf*)&A[l16][k0 + 16 + half * 8];
            #pragma unroll
            for (int q = 0; q < 2; ++q) {
                const int n = (2 * wave + q) * 16 + l16;
                const bf16* pb = BT + (size_t)n * HID + k0 + half * 8;
                FragBF b;
                b.h[0] = *(const v8bf*)(pb);
                b.h[1] = *(const v8bf*)(pb + 16);
                acc[q] = __builtin_amdgcn_wmma_f32_16x16x32_bf16(
                    false, a.v, false, b.v, (short)0, acc[q], false, false);
            }
        }
    };

    v8f yacc[2];   // f32 ODE state for this wave's 32 columns

    for (int t = 0; t < TSTEPS; ++t) {
        // ---- z = h@Wh (+xproj later): 8 accum tiles = cols j, j+256, j+512, j+768
        v8f zacc[8];
        #pragma unroll
        for (int i = 0; i < 8; ++i) zacc[i] = (v8f){0.f,0.f,0.f,0.f,0.f,0.f,0.f,0.f};
        for (int k0 = 0; k0 < HID; k0 += 32) {
            FragBF a;
            a.h[0] = *(const v8bf*)&hbuf[l16][k0 + half * 8];
            a.h[1] = *(const v8bf*)&hbuf[l16][k0 + 16 + half * 8];
            #pragma unroll
            for (int gr = 0; gr < 4; ++gr) {
                #pragma unroll
                for (int q = 0; q < 2; ++q) {
                    const int n = gr * HID + (2 * wave + q) * 16 + l16;
                    const bf16* pb = WhT + (size_t)n * HID + k0 + half * 8;
                    FragBF b;
                    b.h[0] = *(const v8bf*)(pb);
                    b.h[1] = *(const v8bf*)(pb + 16);
                    zacc[gr * 2 + q] = __builtin_amdgcn_wmma_f32_16x16x32_bf16(
                        false, a.v, false, b.v, (short)0, zacc[gr * 2 + q], false, false);
                }
            }
        }
        __syncthreads();   // all waves done reading hbuf

        // ---- LSTM gates (register-resident quadruples + xproj from global)
        #pragma unroll
        for (int q = 0; q < 2; ++q) {
            const int j = (2 * wave + q) * 16 + l16;
            #pragma unroll
            for (int v = 0; v < 8; ++v) {
                const int m = v + 8 * half;
                const size_t xb = ((size_t)(bbase + m) * TSTEPS + t) * (4 * HID);
                float ig = sigf (zacc[0 + q][v] + xproj[xb + j]);
                float fg = sigf (zacc[2 + q][v] + xproj[xb + HID + j]);
                float gg = tanhf(zacc[4 + q][v] + xproj[xb + 2 * HID + j]);
                float og = sigf (zacc[6 + q][v] + xproj[xb + 3 * HID + j]);
                float cn = fg * cstate[m][j] + ig * gg;
                float hn = og * tanhf(cn);
                cstate[m][j] = cn;
                hbuf[m][j]  = (bf16)hn;
                abufA[m][j] = (bf16)hn;
                yacc[q][v] = hn;
            }
        }
        __syncthreads();   // abufA (= y bf16) ready

        // ---- ODE chain: 9 predictions x 3 Euler steps, all in-LDS
        for (int p = 0; p < PREDN; ++p) {
            for (int e = 0; e < 3; ++e) {
                // u = tanh(y@Wf1 + bf1)  : read abufA, write abufB
                v8f ua[2];
                #pragma unroll
                for (int q = 0; q < 2; ++q) ua[q] = (v8f){0.f,0.f,0.f,0.f,0.f,0.f,0.f,0.f};
                kloop2(abufA, Wf1T, ua);
                #pragma unroll
                for (int q = 0; q < 2; ++q) {
                    const int j = (2 * wave + q) * 16 + l16;
                    const float bn = bf1[j];
                    #pragma unroll
                    for (int v = 0; v < 8; ++v) {
                        const int m = v + 8 * half;
                        abufB[m][j] = (bf16)tanhf(ua[q][v] + bn);
                    }
                }
                __syncthreads();   // abufB (= u bf16) ready

                // y += dt*(u@Wf2 + bf2) : read abufB, write abufA
                v8f va[2];
                #pragma unroll
                for (int q = 0; q < 2; ++q) va[q] = (v8f){0.f,0.f,0.f,0.f,0.f,0.f,0.f,0.f};
                kloop2(abufB, Wf2T, va);
                #pragma unroll
                for (int q = 0; q < 2; ++q) {
                    const int j = (2 * wave + q) * 16 + l16;
                    const float bn = bf2[j];
                    #pragma unroll
                    for (int v = 0; v < 8; ++v) {
                        const int m = v + 8 * half;
                        yacc[q][v] += dt * (va[q][v] + bn);
                        abufA[m][j] = (bf16)yacc[q][v];
                    }
                }
                __syncthreads();   // abufA (= new y bf16) ready
            }

            // ---- logits = y@Wfc + bfc  (4 tiles; waves 0..3)
            if (wave < 4) {
                v8f la = {0.f,0.f,0.f,0.f,0.f,0.f,0.f,0.f};
                for (int k0 = 0; k0 < HID; k0 += 32) {
                    FragBF a;
                    a.h[0] = *(const v8bf*)&abufA[l16][k0 + half * 8];
                    a.h[1] = *(const v8bf*)&abufA[l16][k0 + 16 + half * 8];
                    const int n = wave * 16 + l16;
                    const bf16* pb = WfcT + (size_t)n * HID + k0 + half * 8;
                    FragBF b;
                    b.h[0] = *(const v8bf*)(pb);
                    b.h[1] = *(const v8bf*)(pb + 16);
                    la = __builtin_amdgcn_wmma_f32_16x16x32_bf16(
                        false, a.v, false, b.v, (short)0, la, false, false);
                }
                const int n = wave * 16 + l16;
                const float bn = bfc[n];
                #pragma unroll
                for (int v = 0; v < 8; ++v)
                    lbuf[v + 8 * half][n] = la[v] + bn;
            }
            __syncthreads();   // logits ready

            // ---- softmax over 64 + store to out[b, t*9+p, :]
            if (tid < 16) {
                float mx = lbuf[tid][0];
                for (int c0 = 1; c0 < OUTF; ++c0) mx = fmaxf(mx, lbuf[tid][c0]);
                float s = 0.f;
                for (int c0 = 0; c0 < OUTF; ++c0) s += expf(lbuf[tid][c0] - mx);
                srow[tid][0] = mx;
                srow[tid][1] = 1.f / s;
            }
            __syncthreads();
            {
                const int row = tid >> 4;
                const int col = (tid & 15) * 4;
                const float mx = srow[row][0], inv = srow[row][1];
                float* dst = out + ((size_t)(bbase + row) * (TSTEPS * PREDN)
                                    + (t * PREDN + p)) * OUTF + col;
                #pragma unroll
                for (int i = 0; i < 4; ++i)
                    dst[i] = expf(lbuf[row][col + i] - mx) * inv;
            }
            __syncthreads();   // lbuf reusable next p
        }
    }
}

// ---------------------------------------------------------------------------
// Fill / transpose / stats / conv-front-end helpers (unchanged from round 1)
// ---------------------------------------------------------------------------
__global__ void fill_f32(float* p, long n)
{
    long i = (long)blockIdx.x * blockDim.x + threadIdx.x;
    if (i < n) p[i] = 0.f;
}

__global__ void transpose_cvt(const float* __restrict__ src, bf16* __restrict__ dst,
                              int K, int N)
{
    long idx = (long)blockIdx.x * blockDim.x + threadIdx.x;
    if (idx >= (long)K * N) return;
    int  k = (int)(idx % K);
    long n = idx / K;
    dst[idx] = (bf16)src[(long)k * N + n];
}

__global__ void channel_stats(const float* __restrict__ src, long n_elems, int C,
                              float* __restrict__ stats)
{
    __shared__ float ssum[256];
    __shared__ float ssq[256];
    const int tid = threadIdx.x;
    ssum[tid] = 0.f; ssq[tid] = 0.f;
    __syncthreads();

    const long stride = (long)gridDim.x * blockDim.x;   // multiple of C
    long i0 = (long)blockIdx.x * blockDim.x + tid;
    const int c = (int)(i0 % C);
    float s = 0.f, q = 0.f;
    for (long i = i0; i < n_elems; i += stride) {
        float v = src[i];
        s += v; q += v * v;
    }
    atomicAdd(&ssum[c], s);
    atomicAdd(&ssq[c], q);
    __syncthreads();
    if (tid < C) {
        atomicAdd(&stats[tid],     ssum[tid]);
        atomicAdd(&stats[C + tid], ssq[tid]);
    }
}

__global__ void conv1_kernel(const float* __restrict__ x, const float* __restrict__ k1,
                             const float* __restrict__ b1, const float* __restrict__ stats0,
                             float* __restrict__ out)
{
    long idx = (long)blockIdx.x * blockDim.x + threadIdx.x;
    const long total = (long)BATCH * TSTEPS * W1 * 64;
    if (idx >= total) return;
    int  co = (int)(idx % 64);
    long r  = idx / 64;
    int  ow = (int)(r % W1);
    long bt = r / W1;

    const float N0 = (float)((long)BATCH * TSTEPS * WIDTH0);
    float m0 = stats0[0] / N0, m1 = stats0[1] / N0;
    float v0 = stats0[2] / N0 - m0 * m0;
    float v1 = stats0[3] / N0 - m1 * m1;
    float rs0 = rsqrtf(v0 + EPSV), rs1 = rsqrtf(v1 + EPSV);

    float acc = b1[co];
    #pragma unroll
    for (int kw = 0; kw < 3; ++kw) {
        int iw = 3 * ow + kw - 1;                 // pad_left = 1
        if (iw < 0 || iw >= WIDTH0) continue;
        const float* px = x + ((bt * WIDTH0) + iw) * 2;
        acc += (px[0] - m0) * rs0 * k1[(kw * 2 + 0) * 64 + co];
        acc += (px[1] - m1) * rs1 * k1[(kw * 2 + 1) * 64 + co];
    }
    out[idx] = acc;
}

__global__ void im2col_nr(const float* __restrict__ src, const float* __restrict__ stats,
                          bf16* __restrict__ dst, int Win, int Wout, int Cin, float Ncnt)
{
    const int Kc = 3 * Cin;
    long idx = (long)blockIdx.x * blockDim.x + threadIdx.x;
    const long total = (long)BATCH * TSTEPS * Wout * Kc;
    if (idx >= total) return;
    int  j  = (int)(idx % Kc);
    long r  = idx / Kc;
    int  ow = (int)(r % Wout);
    long bt = r / Wout;
    int  kw = j / Cin;
    int  ci = j - kw * Cin;
    int  iw = 3 * ow + kw;                        // pad_left = 0

    float val = 0.f;
    if (iw < Win) {
        float a = src[((bt * Win) + iw) * (long)Cin + ci];
        float m = stats[ci] / Ncnt;
        float v = stats[Cin + ci] / Ncnt - m * m;
        a = (a - m) * rsqrtf(v + EPSV);
        val = fmaxf(a, 0.f);
    }
    dst[idx] = (bf16)val;
}

__global__ void feats_kernel(const float* __restrict__ src, const float* __restrict__ stats,
                             bf16* __restrict__ dst)
{
    long idx = (long)blockIdx.x * blockDim.x + threadIdx.x;
    const long total = (long)BATCH * TSTEPS * HID;
    if (idx >= total) return;
    int  co = (int)(idx % HID);
    long bt = idx / HID;

    const float Ncnt = (float)((long)BATCH * TSTEPS * W3);
    float m = stats[co] / Ncnt;
    float v = stats[HID + co] / Ncnt - m * m;
    float rs = rsqrtf(v + EPSV);

    float s = 0.f;
    #pragma unroll
    for (int ow = 0; ow < W3; ++ow) {
        float a = (src[((bt * W3) + ow) * (long)HID + co] - m) * rs;
        s += fmaxf(a, 0.f);
    }
    dst[idx] = (bf16)(s * (1.0f / (float)W3));
}

// ---------------------------------------------------------------------------
// Host orchestration
// ---------------------------------------------------------------------------
static inline void launch_gemm(hipStream_t s, const bf16* A, const bf16* BT,
                               const float* Ci, int ldCi, const float* bias,
                               float* Co, int ldCo, int M, int N, int K)
{
    dim3 blk(32, 4), grd((N + 63) / 64, M / 16);
    hipLaunchKernelGGL(gemm_bf16_wmma, grd, blk, 0, s,
                       A, BT, Ci, ldCi, bias, Co, ldCo, M, N, K);
}

extern "C" void kernel_launch(void* const* d_in, const int* in_sizes, int n_in,
                              void* d_out, int out_size, void* d_ws, size_t ws_size,
                              hipStream_t stream)
{
    (void)in_sizes; (void)n_in; (void)out_size; (void)ws_size;

    const float* x    = (const float*)d_in[0];
    const float* k1   = (const float*)d_in[1];
    const float* b1   = (const float*)d_in[2];
    const float* k2   = (const float*)d_in[3];
    const float* b2   = (const float*)d_in[4];
    const float* k3   = (const float*)d_in[5];
    const float* b3   = (const float*)d_in[6];
    const float* Wx   = (const float*)d_in[7];
    const float* Wh   = (const float*)d_in[8];
    const float* blst = (const float*)d_in[9];
    const float* Wf1  = (const float*)d_in[10];
    const float* bf1  = (const float*)d_in[11];
    const float* Wf2  = (const float*)d_in[12];
    const float* bf2  = (const float*)d_in[13];
    const float* Wfc  = (const float*)d_in[14];
    const float* bfc  = (const float*)d_in[15];
    float* out = (float*)d_out;

    char* ws = (char*)d_ws;
    size_t off = 0;
    auto alloc = [&](size_t bytes) -> char* {
        off = (off + 255) & ~(size_t)255;
        char* p = ws + off; off += bytes; return p;
    };

    float* stats   = (float*)alloc(4 * 512 * sizeof(float));
    bf16*  WxT     = (bf16*) alloc((size_t)1024 * 256 * 2);
    bf16*  WhT     = (bf16*) alloc((size_t)1024 * 256 * 2);
    bf16*  Wf1T    = (bf16*) alloc((size_t)256 * 256 * 2);
    bf16*  Wf2T    = (bf16*) alloc((size_t)256 * 256 * 2);
    bf16*  WfcT    = (bf16*) alloc((size_t)64 * 256 * 2);
    bf16*  k2T     = (bf16*) alloc((size_t)256 * 192 * 2);
    bf16*  k3T     = (bf16*) alloc((size_t)256 * 768 * 2);
    bf16*  featsbf = (bf16*) alloc((size_t)2560 * 256 * 2);
    float* xproj   = (float*)alloc((size_t)2560 * 1024 * 4);
    char*  slabA   = alloc((size_t)74240 * 256 * 4);   // conv1_raw / conv2_raw / conv3_raw
    char*  slabB   = alloc((size_t)25600 * 768 * 2);   // im2col2 / im2col3

    float* conv1_raw = (float*)slabA;
    bf16*  im2col2   = (bf16*)slabB;
    float* conv2_raw = (float*)slabA;
    bf16*  im2col3   = (bf16*)slabB;
    float* conv3_raw = (float*)slabA;

    float* st0 = stats;        // C=2
    float* st1 = stats + 512;  // C=64
    float* st2 = stats + 1024; // C=256
    float* st3 = stats + 1536; // C=256

    // --- init stats accumulators
    hipLaunchKernelGGL(fill_f32, dim3(8), dim3(256), 0, stream, stats, (long)2048);

    // --- weight transposes -> bf16 (BT layout for WMMA B operand)
    hipLaunchKernelGGL(transpose_cvt, dim3((1024*256+255)/256), dim3(256), 0, stream, Wx,  WxT,  256, 1024);
    hipLaunchKernelGGL(transpose_cvt, dim3((1024*256+255)/256), dim3(256), 0, stream, Wh,  WhT,  256, 1024);
    hipLaunchKernelGGL(transpose_cvt, dim3((256*256+255)/256),  dim3(256), 0, stream, Wf1, Wf1T, 256, 256);
    hipLaunchKernelGGL(transpose_cvt, dim3((256*256+255)/256),  dim3(256), 0, stream, Wf2, Wf2T, 256, 256);
    hipLaunchKernelGGL(transpose_cvt, dim3((256*64+255)/256),   dim3(256), 0, stream, Wfc, WfcT, 256, 64);
    hipLaunchKernelGGL(transpose_cvt, dim3((192*256+255)/256),  dim3(256), 0, stream, k2,  k2T,  192, 256);
    hipLaunchKernelGGL(transpose_cvt, dim3((768*256+255)/256),  dim3(256), 0, stream, k3,  k3T,  768, 256);

    // --- conv front-end ---------------------------------------------------
    hipLaunchKernelGGL(channel_stats, dim3(1024), dim3(256), 0, stream,
                       x, (long)BATCH * TSTEPS * WIDTH0 * 2, 2, st0);
    {
        long total = (long)BATCH * TSTEPS * W1 * 64;
        hipLaunchKernelGGL(conv1_kernel, dim3((unsigned)((total + 255) / 256)), dim3(256),
                           0, stream, x, k1, b1, st0, conv1_raw);
    }
    hipLaunchKernelGGL(channel_stats, dim3(1024), dim3(256), 0, stream,
                       conv1_raw, (long)BATCH * TSTEPS * W1 * 64, 64, st1);
    {
        long total = (long)BATCH * TSTEPS * W2 * 192;
        hipLaunchKernelGGL(im2col_nr, dim3((unsigned)((total + 255) / 256)), dim3(256),
                           0, stream, conv1_raw, st1, im2col2, W1, W2, 64,
                           (float)((long)BATCH * TSTEPS * W1));
    }
    launch_gemm(stream, im2col2, k2T, nullptr, 0, b2, conv2_raw, 256,
                BATCH * TSTEPS * W2, 256, 192);
    hipLaunchKernelGGL(channel_stats, dim3(1024), dim3(256), 0, stream,
                       conv2_raw, (long)BATCH * TSTEPS * W2 * 256, 256, st2);
    {
        long total = (long)BATCH * TSTEPS * W3 * 768;
        hipLaunchKernelGGL(im2col_nr, dim3((unsigned)((total + 255) / 256)), dim3(256),
                           0, stream, conv2_raw, st2, im2col3, W2, W3, 256,
                           (float)((long)BATCH * TSTEPS * W2));
    }
    launch_gemm(stream, im2col3, k3T, nullptr, 0, b3, conv3_raw, 256,
                BATCH * TSTEPS * W3, 256, 768);
    hipLaunchKernelGGL(channel_stats, dim3(1024), dim3(256), 0, stream,
                       conv3_raw, (long)BATCH * TSTEPS * W3 * 256, 256, st3);
    hipLaunchKernelGGL(feats_kernel, dim3(2560), dim3(256), 0, stream,
                       conv3_raw, st3, featsbf);

    // --- x-projection for all timesteps: (2560x256)@(256x1024) + blstm
    launch_gemm(stream, featsbf, WxT, nullptr, 0, blst, xproj, 1024,
                2560, 1024, 256);

    // --- entire recurrence (LSTM + ODE + softmax) in ONE kernel -----------
    const float dt = (1.0f / (float)(PREDN + 1)) / 3.0f;
    hipLaunchKernelGGL(recurrent_fused, dim3(BATCH / 16), dim3(256), 0, stream,
                       xproj, WhT, Wf1T, Wf2T, WfcT, bf1, bf2, bfc, out, dt);
}
